// Robust_AMD_INT8_14748917694596
// MI455X (gfx1250) — matrix operand order repaired and checked by reading, not compiled, not run on hardware
//
#include <hip/hip_runtime.h>
#include <math.h>

// ---------------------------------------------------------------------------
// Fused INT8-dequant VAE/MLP for MI455X (gfx1250), wave32 + WMMA f16.
//
// v3: LDS activation buffers padded (+8 halves per row) so WMMA A-fragment
// ds_load_b128 across 16 rows hits 16 distinct bank groups (stride/2 mod 64
// generates a full residue set) -- removes the 4-way (608) and 16-way (384)
// bank conflicts of v2. Strides stay multiples of 8 halves (16B alignment).
// ---------------------------------------------------------------------------

typedef __attribute__((ext_vector_type(16))) _Float16 v16h;
typedef __attribute__((ext_vector_type(8)))  _Float16 v8h;
typedef __attribute__((ext_vector_type(8)))  float    v8f;

struct Params {
    const _Float16* W[10];   // dequantized padded weights [Npad][Kpad]
    const float*    b[10];   // padded biases [Npad]
    const float*    x;       // [B,379]
    const float*    eps;     // [B,80]
    float* out_mlp;          // [B,2]
    float* out_vae;          // [B,379]
    float* out_mu;           // [B,80]
    float* out_sigma;        // [B,80]
};

#define TM      64
#define NWAVES  8

// Conflict-free LDS row strides (halves): stride/2 mod 64 has order >= 16.
#define S_BIG   616     // for 608-wide buffers   (308 dw, 52*r distinct)
#define S_X     392     // for 384-wide x staging (196 dw,  4*r distinct)
#define S_Z     104     // for 96-wide z          ( 52 dw)
#define S_M     168     // for 160-wide m         ( 84 dw)
#define S_T     72      // for 64-wide t1/t2      ( 36 dw)

enum { ACT_NONE = 0, ACT_ELU, ACT_TANH, ACT_SOFTPLUS, ACT_SIGSP };
enum { OUT_LDS16 = 0, OUT_LDS32, OUT_GLOBAL };

__device__ __forceinline__ float softplus_f(float x) {
    return x > 0.f ? x + log1pf(expf(-x)) : log1pf(expf(x));
}

template <int ACT>
__device__ __forceinline__ float act_f(float x) {
    if constexpr (ACT == ACT_ELU)           return x > 0.f ? x : expm1f(x);
    else if constexpr (ACT == ACT_TANH)     return tanhf(x);
    else if constexpr (ACT == ACT_SOFTPLUS) return softplus_f(x);
    else if constexpr (ACT == ACT_SIGSP) {
        float s = softplus_f(x);
        return 1.f / (1.f + expf(-s));
    } else return x;
}

// A fragment: lane holds row m; lanes 0-15 -> K 0..15 ; 16-31 -> 16..31
__device__ __forceinline__ v16h load_a(const _Float16* arow, int k0) {
    v8h lo = *(const v8h*)(arow + k0);
    v8h hi = *(const v8h*)(arow + k0 + 8);
    return __builtin_shufflevector(lo, hi,
             0, 1, 2, 3, 4, 5, 6, 7, 8, 9, 10, 11, 12, 13, 14, 15);
}

// Store one 16x16 f32 D tile through bias+activation.
template <int ACT, int OUT>
__device__ __forceinline__ void store_tile(const v8f& c, int m0, int col, int g,
                                           const float* bias, void* outp,
                                           int strideOut, int validN) {
    const float bb = bias[col];
#pragma unroll
    for (int v = 0; v < 8; ++v) {
        const int m = m0 + v + 8 * g;           // D row: VGPR v + 8*lanegroup
        float xv = c[v] + bb;
        if constexpr (OUT == OUT_LDS16) {
            ((_Float16*)outp)[m * strideOut + col] = (_Float16)act_f<ACT>(xv);
        } else if constexpr (OUT == OUT_LDS32) {
            ((float*)outp)[m * strideOut + col] = xv;
        } else {
            if (col < validN)
                ((float*)outp)[(size_t)m * strideOut + col] = act_f<ACT>(xv);
        }
    }
}

// One linear layer on a 64-row LDS tile; each wave owns 32x32 output supertiles.
//   in : LDS f16 activations [TM][strideIn], K%32==0, strideIn%8==0
//   W  : global f16 [Npad][K] (scale folded), bias: global f32 [Npad], N%32==0
template <int ACT, int OUT>
__device__ __forceinline__ void layer(const _Float16* __restrict__ W,
                                      const float* __restrict__ bias,
                                      const _Float16* __restrict__ in,
                                      int K, int strideIn, int N,
                                      void* outp, int strideOut, int validN,
                                      int tid)
{
    const int wave = tid >> 5;
    const int lane = tid & 31;
    const int l15  = lane & 15;
    const int g    = lane >> 4;
    const int nt   = N >> 5;              // 32-wide N supertiles
    const int tiles = (TM / 32) * nt;     // 2 M supertiles

    for (int t = wave; t < tiles; t += NWAVES) {
        const int m0 = (t / nt) << 5;
        const int n0 = (t % nt) << 5;

        v8f c00 = {}, c01 = {}, c10 = {}, c11 = {};
        const _Float16* arow0 = in + (m0 + l15) * strideIn + g * 16;
        const _Float16* arow1 = arow0 + 16 * strideIn;
        // B: lane holds col; lanes 0-15 -> K 0..15 ; lanes 16-31 -> K 16..31
        const _Float16* brow0 = W + (size_t)(n0 + l15) * K + g * 16;
        const _Float16* brow1 = brow0 + (size_t)16 * K;

#pragma unroll 2
        for (int k0 = 0; k0 < K; k0 += 32) {
            v16h a0 = load_a(arow0, k0);
            v16h a1 = load_a(arow1, k0);
            v16h b0 = *(const v16h*)(brow0 + k0);   // 32B contiguous per lane
            v16h b1 = *(const v16h*)(brow1 + k0);
            c00 = __builtin_amdgcn_wmma_f32_16x16x32_f16(false, a0, false, b0, (short)0, c00, false, false);
            c01 = __builtin_amdgcn_wmma_f32_16x16x32_f16(false, a0, false, b1, (short)0, c01, false, false);
            c10 = __builtin_amdgcn_wmma_f32_16x16x32_f16(false, a1, false, b0, (short)0, c10, false, false);
            c11 = __builtin_amdgcn_wmma_f32_16x16x32_f16(false, a1, false, b1, (short)0, c11, false, false);
        }

        const int col0 = n0 + l15;
        const int col1 = col0 + 16;
        store_tile<ACT, OUT>(c00, m0,      col0, g, bias, outp, strideOut, validN);
        store_tile<ACT, OUT>(c01, m0,      col1, g, bias, outp, strideOut, validN);
        store_tile<ACT, OUT>(c10, m0 + 16, col0, g, bias, outp, strideOut, validN);
        store_tile<ACT, OUT>(c11, m0 + 16, col1, g, bias, outp, strideOut, validN);
    }
}

// ---------------------------------------------------------------------------
// Dequantize one layer: Wf[r*Kpad+k] = Wq[r*n+k] * s[r]/127 (0-padded),
// plus padded bias. One launch per layer.
// ---------------------------------------------------------------------------
extern "C" __global__ void dequant_kernel(const signed char* __restrict__ Wq,
                                          const float* __restrict__ s,
                                          const float* __restrict__ b,
                                          int o, int n, int Npad, int Kpad,
                                          _Float16* __restrict__ Wf,
                                          float* __restrict__ bp)
{
    const int idx = blockIdx.x * blockDim.x + threadIdx.x;
    const int total = Npad * Kpad;
    if (idx < total) {
        const int r = idx / Kpad;
        const int k = idx - r * Kpad;
        float v = 0.f;
        if (r < o && k < n) v = (float)Wq[r * n + k] * (s[r] * (1.f / 127.f));
        Wf[idx] = (_Float16)v;
    }
    if (idx < Npad) bp[idx] = (idx < o) ? b[idx] : 0.f;
}

// ---------------------------------------------------------------------------
// Fused forward pass: one 64-row tile per workgroup, everything in LDS.
// Dynamic LDS: buf0(64x616 f16) + buf1(64x616 f16, reused as f32 h2)
//              + aux: z(64x104) m(64x168) t1(64x72) t2(64x72)  = 210944 B
// ---------------------------------------------------------------------------
extern "C" __global__ __launch_bounds__(256)
void fused_vae_kernel(Params P)
{
    extern __shared__ char smem[];
    constexpr int BUFH = TM * S_BIG;
    _Float16* buf0 = (_Float16*)smem;
    _Float16* buf1 = buf0 + BUFH;
    float*    h2   = (float*)buf1;          // 64x160 f32 fits in buf1 (40KB<78.8KB)
    _Float16* zb   = buf1 + BUFH;           // 64 x S_Z
    _Float16* mb   = zb + TM * S_Z;         // 64 x S_M
    _Float16* t1   = mb + TM * S_M;         // 64 x S_T
    _Float16* t2   = t1 + TM * S_T;         // 64 x S_T

    const int tid  = threadIdx.x;
    const int row0 = blockIdx.x * TM;

    // Zero aux region once (K-padding zeros for z/m/t buffers).
    for (int i = tid; i < TM * (S_Z + S_M + 2 * S_T); i += 256)
        zb[i] = (_Float16)0.f;

    // Stage x tile -> f16 LDS [64][S_X], zero-padded cols 379..383.
    for (int i = tid; i < TM * 384; i += 256) {
        const int m = i / 384, c2 = i - m * 384;
        float v = (c2 < 379) ? P.x[(size_t)(row0 + m) * 379 + c2] : 0.f;
        buf0[m * S_X + c2] = (_Float16)v;
    }
    __syncthreads();

    // Encoder
    layer<ACT_ELU,  OUT_LDS16>(P.W[0], P.b[0], buf0, 384, S_X,   608, buf1, S_BIG, 608, tid);
    __syncthreads();
    layer<ACT_TANH, OUT_LDS16>(P.W[1], P.b[1], buf1, 608, S_BIG, 608, buf0, S_BIG, 608, tid);
    __syncthreads();
    layer<ACT_NONE, OUT_LDS32>(P.W[2], P.b[2], buf0, 608, S_BIG, 160, h2,   160,   160, tid);
    __syncthreads();

    // Split: mu/sigma outputs, z = fp16(mu + sigma*eps), m = [mu|sigma] (f16).
    for (int i = tid; i < TM * 80; i += 256) {
        const int m = i / 80, n = i - m * 80;
        const size_t row = (size_t)(row0 + m);
        const float mu = h2[m * 160 + n];
        const float sg = softplus_f(h2[m * 160 + 80 + n]);
        P.out_mu[row * 80 + n]    = mu;
        P.out_sigma[row * 80 + n] = sg;
        const float e = P.eps[row * 80 + n];
        zb[m * S_Z + n]      = (_Float16)(mu + sg * e);   // fp16 round-trip
        mb[m * S_M + n]      = (_Float16)mu;
        mb[m * S_M + 80 + n] = (_Float16)sg;
    }
    __syncthreads();

    // Decoder
    layer<ACT_TANH,  OUT_LDS16>(P.W[3], P.b[3], zb,   96,  S_Z,   608, buf0, S_BIG, 608, tid);
    __syncthreads();
    layer<ACT_ELU,   OUT_LDS16>(P.W[4], P.b[4], buf0, 608, S_BIG, 608, buf1, S_BIG, 608, tid);
    __syncthreads();
    layer<ACT_SIGSP, OUT_GLOBAL>(P.W[5], P.b[5], buf1, 608, S_BIG, 384,
                                 P.out_vae + (size_t)row0 * 379, 379, 379, tid);
    __syncthreads();

    // MLP head on m = [mu|sigma]
    layer<ACT_TANH, OUT_LDS16>(P.W[6], P.b[6], mb, 160, S_M, 64, t1, S_T, 64, tid);
    __syncthreads();
    layer<ACT_ELU,  OUT_LDS16>(P.W[7], P.b[7], t1,  64,  S_T, 64, t2, S_T, 64, tid);
    __syncthreads();
    layer<ACT_ELU,  OUT_LDS16>(P.W[8], P.b[8], t2,  64,  S_T, 64, t1, S_T, 64, tid);
    __syncthreads();
    layer<ACT_SOFTPLUS, OUT_GLOBAL>(P.W[9], P.b[9], t1, 64, S_T, 32,
                                    P.out_mlp + (size_t)row0 * 2, 2, 2, tid);
}

// ---------------------------------------------------------------------------
// Host side
// ---------------------------------------------------------------------------
extern "C" void kernel_launch(void* const* d_in, const int* in_sizes, int n_in,
                              void* d_out, int out_size, void* d_ws, size_t ws_size,
                              hipStream_t stream)
{
    static const int O[10]  = {600, 600, 160, 600, 600, 379, 40, 40, 40, 2};
    static const int NI[10] = {379, 600, 600,  80, 600, 600, 160, 40, 40, 40};

    int Np[10], Kp[10];
    size_t woff[10], off = 0;
    for (int i = 0; i < 10; ++i) {
        Np[i] = ((O[i] + 31) / 32) * 32;     // N padded to 32 for supertiles
        Kp[i] = ((NI[i] + 31) / 32) * 32;
        woff[i] = off;
        off += (size_t)Np[i] * Kp[i];
    }

    _Float16* wbase = (_Float16*)d_ws;
    float*    bbase = (float*)((char*)d_ws + off * sizeof(_Float16));

    Params P;
    size_t boff = 0;
    for (int i = 0; i < 10; ++i) {
        _Float16* Wf = wbase + woff[i];
        float*    bp = bbase + boff;
        P.W[i] = Wf;
        P.b[i] = bp;
        const int total = Np[i] * Kp[i];
        dequant_kernel<<<(total + 255) / 256, 256, 0, stream>>>(
            (const signed char*)d_in[2 + 3 * i],
            (const float*)d_in[3 + 3 * i],
            (const float*)d_in[4 + 3 * i],
            O[i], NI[i], Np[i], Kp[i], Wf, bp);
        boff += Np[i];
    }

    const int Bsz = in_sizes[0] / 379;     // 131072
    P.x   = (const float*)d_in[0];
    P.eps = (const float*)d_in[1];
    float* out = (float*)d_out;
    P.out_mlp   = out;                                  // [B,2]
    P.out_vae   = out + (size_t)Bsz * 2;                // [B,379]
    P.out_mu    = out + (size_t)Bsz * (2 + 379);        // [B,80]
    P.out_sigma = P.out_mu + (size_t)Bsz * 80;          // [B,80]

    const size_t shbytes = (size_t)(TM * S_BIG) * 2 * sizeof(_Float16)   // buf0+buf1
                         + (size_t)TM * (S_Z + S_M + 2 * S_T) * sizeof(_Float16); // aux
    // 210944 B of dynamic LDS (<= 320KB/WGP on CDNA5); raise the cap.
    (void)hipFuncSetAttribute(reinterpret_cast<const void*>(fused_vae_kernel),
                              hipFuncAttributeMaxDynamicSharedMemorySize,
                              (int)shbytes);

    fused_vae_kernel<<<Bsz / TM, 256, shbytes, stream>>>(P);
}
